// DeepEMD_12163347382842
// MI455X (gfx1250) — matrix operand order repaired and checked by reading, not compiled, not run on hardware
//
#include <hip/hip_runtime.h>
#include <hip/hip_bf16.h>
#include <stdint.h>

// Problem constants (from reference): Q=75, P=5, C=640, H=W=14 -> HW=196
#define Q_   75
#define P_   5
#define C_   640
#define HW_  196
#define HWP_ 208   // HW padded to 13 * 16 for WMMA tiling
#define NT_  13    // 16x16 tiles per (padded) spatial dim
#define EPS_ 1e-8f

typedef __attribute__((ext_vector_type(16))) __bf16        v16bf;
typedef __attribute__((ext_vector_type(8)))  float         v8f;
typedef __attribute__((ext_vector_type(4)))  unsigned int  u32x4;
typedef __attribute__((ext_vector_type(8)))  unsigned int  u32x8;

__device__ __forceinline__ unsigned short f32_to_bf16(float x) {
    unsigned int u = __builtin_bit_cast(unsigned int, x);
    u += 0x7FFFu + ((u >> 16) & 1u);        // round-to-nearest-even
    return (unsigned short)(u >> 16);
}

__device__ __forceinline__ float wave32_allreduce_sum(float v) {
    #pragma unroll
    for (int m = 16; m >= 1; m >>= 1) v += __shfl_xor(v, m, 32);
    return v;
}

// ---------------------------------------------------------------------------
// K1: adaptive_avg_pool2d means: supPool[p][c], qryPool[q][c]. One wave per
// (image, channel); lanes stride the 196 contiguous pixels (coalesced).
// ---------------------------------------------------------------------------
__global__ void pool_kernel(const float* __restrict__ sup, const float* __restrict__ qry,
                            float* __restrict__ supPool, float* __restrict__ qryPool) {
    int wid  = (int)((blockIdx.x * blockDim.x + threadIdx.x) >> 5);
    int lane = threadIdx.x & 31;
    if (wid >= (P_ + Q_) * C_) return;
    const float* src; float* dst;
    if (wid < P_ * C_) { src = sup + (size_t)wid * HW_;              dst = supPool + wid; }
    else { int j = wid - P_ * C_; src = qry + (size_t)j * HW_;       dst = qryPool + j;  }
    float s = 0.f;
    for (int i = lane; i < HW_; i += 32) s += src[i];
    s = wave32_allreduce_sum(s);
    if (lane == 0) *dst = s * (1.0f / HW_);
}

// ---------------------------------------------------------------------------
// K2: per-pixel channel-center + L2-normalize, then store transposed bf16:
//   qnT[q][i][c], pnT[p][i][c]  (pixel-major, channel-contiguous, i padded
//   to 208 with zero rows). One wave per pixel, lanes over channels.
// ---------------------------------------------------------------------------
__global__ void normalize_kernel(const float* __restrict__ sup, const float* __restrict__ qry,
                                 unsigned short* __restrict__ qnT, unsigned short* __restrict__ pnT) {
    int wid  = (int)((blockIdx.x * blockDim.x + threadIdx.x) >> 5);
    int lane = threadIdx.x & 31;
    if (wid >= (Q_ + P_) * HWP_) return;
    const float* src; unsigned short* dst; int i;
    if (wid < Q_ * HWP_) {
        int n = wid / HWP_; i = wid % HWP_;
        src = qry + (size_t)n * C_ * HW_;
        dst = qnT + ((size_t)n * HWP_ + i) * C_;
    } else {
        int w = wid - Q_ * HWP_; int n = w / HWP_; i = w % HWP_;
        src = sup + (size_t)n * C_ * HW_;
        dst = pnT + ((size_t)n * HWP_ + i) * C_;
    }
    if (i >= HW_) {                          // zero pad rows -> safe WMMA K/M/N pad
        #pragma unroll
        for (int t = 0; t < C_ / 32; ++t) dst[lane + t * 32] = 0;
        return;
    }
    float vals[C_ / 32];
    float s = 0.f;
    #pragma unroll
    for (int t = 0; t < C_ / 32; ++t) {
        float v = src[(size_t)(lane + t * 32) * HW_ + i];
        vals[t] = v; s += v;
    }
    float mean = wave32_allreduce_sum(s) * (1.0f / C_);
    float ss = 0.f;
    #pragma unroll
    for (int t = 0; t < C_ / 32; ++t) { float v = vals[t] - mean; vals[t] = v; ss += v * v; }
    ss = wave32_allreduce_sum(ss);
    float scale = 1.0f / fmaxf(sqrtf(ss), EPS_);
    #pragma unroll
    for (int t = 0; t < C_ / 32; ++t) dst[lane + t * 32] = f32_to_bf16(vals[t] * scale);
}

// ---------------------------------------------------------------------------
// K3a: weight_1[q][p][i] = relu(sum_c qry[q,c,i] * supPool[p,c]) + 0.001
// One thread per (q,i), 5 accumulators -> query read once, coalesced over i.
// ---------------------------------------------------------------------------
__global__ void weight1_kernel(const float* __restrict__ qry, const float* __restrict__ supPool,
                               float* __restrict__ out1) {
    int tid = (int)(blockIdx.x * blockDim.x + threadIdx.x);
    if (tid >= Q_ * HW_) return;
    int i = tid % HW_, q = tid / HW_;
    const float* qbase = qry + (size_t)q * C_ * HW_ + i;
    float acc[P_] = {};
    for (int c = 0; c < C_; ++c) {
        float qv = qbase[(size_t)c * HW_];
        #pragma unroll
        for (int p = 0; p < P_; ++p) acc[p] += qv * supPool[p * C_ + c];  // uniform -> scalar loads
    }
    #pragma unroll
    for (int p = 0; p < P_; ++p) {
        float r = (acc[p] > 0.f ? acc[p] : 0.f) + 0.001f;
        out1[((size_t)(q * P_ + p)) * HW_ + i] = r;
    }
}

// K3b: weight_2[p][q][i] = relu(sum_c sup[p,c,i] * qryPool[q,c]) + 0.001
// support (2.5 MB) is L2/L0 resident; one thread per output, coalesced over i.
__global__ void weight2_kernel(const float* __restrict__ sup, const float* __restrict__ qryPool,
                               float* __restrict__ out2) {
    int tid = (int)(blockIdx.x * blockDim.x + threadIdx.x);
    if (tid >= P_ * Q_ * HW_) return;
    int i = tid % HW_;
    int q = (tid / HW_) % Q_;
    int p = tid / (HW_ * Q_);
    const float* sbase = sup + (size_t)p * C_ * HW_ + i;
    const float* qp    = qryPool + q * C_;
    float acc = 0.f;
    for (int c = 0; c < C_; ++c) acc += sbase[(size_t)c * HW_] * qp[c];
    out2[tid] = (acc > 0.f ? acc : 0.f) + 0.001f;
}

// ---------------------------------------------------------------------------
// K4: similarity_map[q][p][i][j] = qn[q,:,i] . pn[p,:,j] via
// v_wmma_f32_16x16x32_bf16. One wave per 16x16 tile; 13 waves (a full tile
// row) per block so all waves of a block share the same A tile in WGP$.
// Fragment loads are aligned global_load_b128 thanks to the [*, 208, 640]
// bf16 layout. K loop: 640 / 32 = 20 WMMAs per tile.
// Epilogue: interior tile rows (it < 12, i.e. 12/13 of all blocks) take a
// branch-free 8-store clause; only the boundary row masks per-half, where
// the valid row count is a constant (4 for lanes 0-15, 0 for lanes 16-31).
// ---------------------------------------------------------------------------
__global__ __launch_bounds__(NT_ * 32)
void simmap_wmma_kernel(const unsigned short* __restrict__ qnT,
                        const unsigned short* __restrict__ pnT,
                        float* __restrict__ out) {
    int pair = (int)blockIdx.x / NT_;
    int it   = (int)blockIdx.x % NT_;       // tile row (i)
    int q = pair / P_, p = pair % P_;
    int jt   = threadIdx.x >> 5;            // tile col (j) = wave id
    int lane = threadIdx.x & 31;
    int half = lane >> 4;                   // 0: lanes 0-15, 1: lanes 16-31
    int l16  = lane & 15;

    // A row m = lane%16; K chunks [k+half*8, +8) and [k+16+half*8, +8)
    const unsigned short* aRow = qnT + ((size_t)(q * HWP_ + it * 16 + l16)) * C_ + half * 8;
    // B col n = lane%16; K chunks [k+half*8, +8) and [k+16+half*8, +8)
    const unsigned short* bCol = pnT + ((size_t)(p * HWP_ + jt * 16 + l16)) * C_ + half * 8;

    v8f acc = {};
    #pragma unroll 4
    for (int k = 0; k < C_; k += 32) {
        u32x4 a0 = *(const u32x4*)(aRow + k);
        u32x4 a1 = *(const u32x4*)(aRow + k + 16);
        u32x4 b0 = *(const u32x4*)(bCol + k);
        u32x4 b1 = *(const u32x4*)(bCol + k + 16);
        u32x8 au = {a0.x, a0.y, a0.z, a0.w, a1.x, a1.y, a1.z, a1.w};
        u32x8 bu = {b0.x, b0.y, b0.z, b0.w, b1.x, b1.y, b1.z, b1.w};
        v16bf av = __builtin_bit_cast(v16bf, au);
        v16bf bv = __builtin_bit_cast(v16bf, bu);
        acc = __builtin_amdgcn_wmma_f32_16x16x32_bf16(
                  false, av, false, bv, (short)0, acc, false, false);
    }

    // C/D layout: VGPR r -> M = it*16 + half*8 + r, N = jt*16 + lane%16
    int j = jt * 16 + l16;
    if (j >= HW_) return;                   // only jt == 12 drops lanes
    float* obase = out + ((size_t)(q * P_ + p)) * HW_ * HW_
                       + (size_t)(it * 16 + half * 8) * HW_ + j;
    if (it < NT_ - 1) {
        // fast path: all 8 rows valid, unconditional store clause
        #pragma unroll
        for (int r = 0; r < 8; ++r) obase[(size_t)(r * HW_)] = acc[r];
    } else {
        // boundary row: rows 192..195 valid -> half 0 stores 4, half 1 stores 0
        if (half == 0) {
            #pragma unroll
            for (int r = 0; r < 4; ++r) obase[(size_t)(r * HW_)] = acc[r];
        }
    }
}

// ---------------------------------------------------------------------------
extern "C" void kernel_launch(void* const* d_in, const int* in_sizes, int n_in,
                              void* d_out, int out_size, void* d_ws, size_t ws_size,
                              hipStream_t stream) {
    (void)in_sizes; (void)n_in; (void)out_size; (void)ws_size;
    const float* sup = (const float*)d_in[0];   // [P, C, H, W]
    const float* qry = (const float*)d_in[1];   // [Q, C, H, W]

    float* out = (float*)d_out;
    const size_t simElems = (size_t)Q_ * P_ * HW_ * HW_;   // 14,406,000
    float* w1 = out + simElems;                            // [Q, P, HW]
    float* w2 = w1 + (size_t)Q_ * P_ * HW_;                // [P, Q, HW]

    // Workspace layout (~21.5 MB): qnT | pnT | supPool | qryPool
    unsigned short* qnT = (unsigned short*)d_ws;                    // Q*208*640 bf16
    unsigned short* pnT = qnT + (size_t)Q_ * HWP_ * C_;             // P*208*640 bf16
    float* supPool = (float*)(pnT + (size_t)P_ * HWP_ * C_);        // P*C f32
    float* qryPool = supPool + P_ * C_;                             // Q*C f32

    {   // K1: pooled means
        int threads = (P_ + Q_) * C_ * 32;
        pool_kernel<<<(threads + 255) / 256, 256, 0, stream>>>(sup, qry, supPool, qryPool);
    }
    {   // K2: center + normalize + bf16 transpose (with zero padding)
        int threads = (Q_ + P_) * HWP_ * 32;
        normalize_kernel<<<(threads + 255) / 256, 256, 0, stream>>>(sup, qry, qnT, pnT);
    }
    {   // K3: weight vectors
        int n1 = Q_ * HW_;
        weight1_kernel<<<(n1 + 255) / 256, 256, 0, stream>>>(qry, supPool, w1);
        int n2 = P_ * Q_ * HW_;
        weight2_kernel<<<(n2 + 255) / 256, 256, 0, stream>>>(sup, qryPool, w2);
    }
    {   // K4: 375 pair-GEMMs, 13x13 tiles each, bf16 WMMA with f32 accum
        dim3 grid(Q_ * P_ * NT_);
        dim3 block(NT_ * 32);
        simmap_wmma_kernel<<<grid, block, 0, stream>>>(qnT, pnT, out);
    }
}
